// ContrastiveLossModule_42863773614346
// MI455X (gfx1250) — matrix operand order repaired and checked by reading, not compiled, not run on hardware
//
#include <hip/hip_runtime.h>
#include <hip/hip_bf16.h>
#include <math.h>

typedef __attribute__((ext_vector_type(16))) _Float16 v16h;
typedef __attribute__((ext_vector_type(8)))  _Float16 v8h;
typedef __attribute__((ext_vector_type(8)))  float    v8f;

#define B_     16
#define T_     12
#define N_     307   // nodes
#define C_     64    // channels (GEMM K)
#define S_     4     // subgraphs
#define BT     192   // B_*T_ batches
#define NPAD   320   // N padded to 16
#define NTILES 20
#define M_     1228  // S_*N_
#define MPAD   1232  // M padded to 16
#define MTILES 77
#define EMBROWS 1360 // MPAD + 8*16: allows unconditional B prefetch overrun
#define EPSF   1e-12f

// ---------------------------------------------------------------------------
// Kernel 1: node_memory [B,N,T,C] f32 -> nm_h [bt][n(pad 320)][c] f16
// ---------------------------------------------------------------------------
__global__ void prep_nm_kernel(const float* __restrict__ node_memory,
                               _Float16* __restrict__ nm_h) {
  int idx = blockIdx.x * blockDim.x + threadIdx.x;
  if (idx >= BT * NPAD * C_) return;
  int c  = idx & 63;
  int n  = (idx >> 6) % NPAD;
  int bt = idx / (NPAD * C_);
  int b = bt / T_, t = bt % T_;
  float v = 0.0f;
  if (n < N_) v = node_memory[(((size_t)(b * N_ + n)) * T_ + t) * C_ + c];
  nm_h[idx] = (_Float16)v;
}

// ---------------------------------------------------------------------------
// Kernel 2: subgraph [B,N,T,C,S] f32 -> emb_h [bt][m (EMBROWS rows)][c] f16,
// L2-normalized over C. Rows >= MPAD are never written (prefetch slop only).
// ---------------------------------------------------------------------------
__global__ void prep_emb_kernel(const float* __restrict__ sub,
                                _Float16* __restrict__ emb_h) {
  int m  = blockIdx.x;   // 0..MPAD-1
  int bt = blockIdx.y;   // 0..BT-1
  int c  = threadIdx.x;  // 0..63
  int b = bt / T_, t = bt % T_;
  float x = 0.0f;
  if (m < M_) {
    int s = m / N_, n = m % N_;
    x = sub[((((size_t)(b * N_ + n)) * T_ + t) * C_ + c) * S_ + s];
  }
  float ss = x * x;
  #pragma unroll
  for (int o = 16; o > 0; o >>= 1) ss += __shfl_xor(ss, o, 32);
  __shared__ float wsum[2];
  if ((c & 31) == 0) wsum[c >> 5] = ss;
  __syncthreads();
  float norm = sqrtf(wsum[0] + wsum[1]);
  float inv  = 1.0f / fmaxf(norm, EPSF);
  emb_h[((size_t)bt * EMBROWS + m) * C_ + c] = (_Float16)(x * inv);
}

// ---------------------------------------------------------------------------
// Kernel 3: pack masks into zero-padded [NPAD][MPAD] float2 {pos, pos+neg}.
// ---------------------------------------------------------------------------
__global__ void prep_mask_kernel(const float* __restrict__ pos,
                                 const float* __restrict__ neg,
                                 float2* __restrict__ pn) {
  int idx = blockIdx.x * blockDim.x + threadIdx.x;
  if (idx >= NPAD * MPAD) return;
  int m = idx % MPAD;
  int n = idx / MPAD;
  float p = 0.0f, q = 0.0f;
  if (n < N_ && m < M_) {
    p = pos[(size_t)n * M_ + m];
    q = p + neg[(size_t)n * M_ + m];
  }
  pn[idx] = make_float2(p, q);
}

// ---------------------------------------------------------------------------
// Kernel 4: fused GEMM (WMMA f16 -> f32) + exp + masked reductions.
// Block = (bt, 16-row n-tile); 8 waves sweep disjoint M-tiles. Counted loop,
// unconditional 1-deep B-tile prefetch (emb over-allocated), branch-free body,
// EXEC all-1s at every WMMA, deterministic (no atomics).
// ---------------------------------------------------------------------------
__global__ void __launch_bounds__(256)
sim_loss_main_kernel(const _Float16* __restrict__ nm_h,
                     const _Float16* __restrict__ emb_h,
                     const float2* __restrict__ pn,
                     float* __restrict__ numer,
                     float* __restrict__ denom) {
  const int ntile = blockIdx.x;
  const int bt    = blockIdx.y;
  const int n0    = ntile * 16;
  const int lane  = threadIdx.x & 31;
  const int wave  = threadIdx.x >> 5;
  const int half  = lane >> 4;   // lane group (0: lanes 0-15, 1: lanes 16-31)
  const int lcol  = lane & 15;

  // --- A matrix (nm rows), 16-bit A 16x32, contiguous-k order (same as B), two K-blocks (K=64) ---
  // lane row = lcol; element i: K = kb + 16*half + i (contiguous in c)
  const _Float16* nmrow = nm_h + ((size_t)bt * NPAD + n0 + lcol) * C_;
  union Au { v16h v; v8h h[2]; };
  Au a0, a1;
  a0.h[0] = *reinterpret_cast<const v8h*>(nmrow + half * 16);
  a0.h[1] = *reinterpret_cast<const v8h*>(nmrow + 8 + half * 16);
  a1.h[0] = *reinterpret_cast<const v8h*>(nmrow + 32 + half * 16);
  a1.h[1] = *reinterpret_cast<const v8h*>(nmrow + 40 + half * 16);

  // Mask base: rows n0+half*8 .. +7 at constant immediate offsets r*MPAD.
  const float2* pnrow = pn + (size_t)(n0 + half * 8) * MPAD + lcol;

  float nacc[8], dacc[8];
  #pragma unroll
  for (int r = 0; r < 8; ++r) { nacc[r] = 0.0f; dacc[r] = 0.0f; }

  // --- B matrix (emb rows as columns), 32x16 layout:
  // lane col = lcol; element i: K = kb + 16*half + i (contiguous in c)
  const _Float16* er =
      emb_h + ((size_t)bt * EMBROWS + wave * 16 + lcol) * C_ + 16 * half;

  union Bu { v16h v; v8h h[2]; };
  Bu b0, b1, nb0, nb1;
  b0.h[0] = *reinterpret_cast<const v8h*>(er);
  b0.h[1] = *reinterpret_cast<const v8h*>(er + 8);
  b1.h[0] = *reinterpret_cast<const v8h*>(er + 32);
  b1.h[1] = *reinterpret_cast<const v8h*>(er + 40);

  const int nt = (84 - wave) >> 3;     // ceil((MTILES - wave)/8), wave-uniform
  const float2* pp = pnrow + wave * 16;

  for (int i = 0; i < nt; ++i) {       // wave-uniform counted loop
    // Unconditional prefetch of next B tile (over-allocation absorbs overrun;
    // the final prefetch's data is never consumed).
    er += (size_t)8 * 16 * C_;
    nb0.h[0] = *reinterpret_cast<const v8h*>(er);
    nb0.h[1] = *reinterpret_cast<const v8h*>(er + 8);
    nb1.h[0] = *reinterpret_cast<const v8h*>(er + 32);
    nb1.h[1] = *reinterpret_cast<const v8h*>(er + 40);

    v8f acc = {};
    acc = __builtin_amdgcn_wmma_f32_16x16x32_f16(false, a0.v, false, b0.v,
                                                 (short)0, acc, false, false);
    acc = __builtin_amdgcn_wmma_f32_16x16x32_f16(false, a1.v, false, b1.v,
                                                 (short)0, acc, false, false);

    // C/D layout: VGPR r -> row r (lanes 0-15) / row r+8 (lanes 16-31), col=lcol
    #pragma unroll
    for (int r = 0; r < 8; ++r) {
      float2 pq = pp[(size_t)r * MPAD];          // {pos, pos+neg}, zero-padded
      float e = __expf(acc[r] * 2.0f);           // sim / TEMPERATURE(0.5)
      nacc[r] = fmaf(e, pq.x, nacc[r]);
      dacc[r] = fmaf(e, pq.y, dacc[r]);
    }

    pp += 8 * 16;
    b0 = nb0; b1 = nb1;
  }

  // Reduce over the 16 columns within each half-group (fixed order).
  __shared__ float red[8][16][2];
  #pragma unroll
  for (int r = 0; r < 8; ++r) {
    float nv = nacc[r], dv = dacc[r];
    #pragma unroll
    for (int o = 8; o > 0; o >>= 1) {
      nv += __shfl_xor(nv, o, 16);
      dv += __shfl_xor(dv, o, 16);
    }
    if (lcol == 0) {
      red[wave][r + half * 8][0] = nv;
      red[wave][r + half * 8][1] = dv;
    }
  }
  __syncthreads();

  // Fixed-order cross-wave sum; each (bt,n) owned by exactly this block.
  if (threadIdx.x < 16) {
    int row = threadIdx.x;
    int n = n0 + row;
    if (n < N_) {
      float sn = 0.0f, sd = 0.0f;
      #pragma unroll
      for (int w = 0; w < 8; ++w) { sn += red[w][row][0]; sd += red[w][row][1]; }
      numer[(size_t)bt * N_ + n] = sn;
      denom[(size_t)bt * N_ + n] = sd;
    }
  }
}

// ---------------------------------------------------------------------------
// Kernel 5: loss = -mean(log(num / (den + eps)))
// ---------------------------------------------------------------------------
__global__ void final_loss_kernel(const float* __restrict__ numer,
                                  const float* __restrict__ denom,
                                  float* __restrict__ out) {
  const int total = BT * N_;
  float s = 0.0f;
  for (int i = threadIdx.x; i < total; i += blockDim.x)
    s += logf(numer[i] / (denom[i] + EPSF));
  __shared__ float red[256];
  red[threadIdx.x] = s;
  __syncthreads();
  for (int o = 128; o > 0; o >>= 1) {
    if ((int)threadIdx.x < o) red[threadIdx.x] += red[threadIdx.x + o];
    __syncthreads();
  }
  if (threadIdx.x == 0) out[0] = -red[0] / (float)total;
}

// ---------------------------------------------------------------------------
extern "C" void kernel_launch(void* const* d_in, const int* in_sizes, int n_in,
                              void* d_out, int out_size, void* d_ws, size_t ws_size,
                              hipStream_t stream) {
  const float* node_memory = (const float*)d_in[0];
  const float* sub         = (const float*)d_in[1];
  const float* pos         = (const float*)d_in[2];
  const float* neg         = (const float*)d_in[3];

  char* ws = (char*)d_ws;
  size_t off = 0;
  _Float16* emb_h = (_Float16*)(ws + off); off += (size_t)BT * EMBROWS * C_ * sizeof(_Float16);
  _Float16* nm_h  = (_Float16*)(ws + off); off += (size_t)BT * NPAD * C_ * sizeof(_Float16);
  float2*   pn    = (float2*)(ws + off);   off += (size_t)NPAD * MPAD * sizeof(float2);
  float* numer    = (float*)(ws + off);    off += (size_t)BT * N_ * sizeof(float);
  float* denom    = (float*)(ws + off);    off += (size_t)BT * N_ * sizeof(float);

  const int nm_total = BT * NPAD * C_;
  prep_nm_kernel<<<(nm_total + 255) / 256, 256, 0, stream>>>(node_memory, nm_h);
  prep_emb_kernel<<<dim3(MPAD, BT), 64, 0, stream>>>(sub, emb_h);
  const int mk_total = NPAD * MPAD;
  prep_mask_kernel<<<(mk_total + 255) / 256, 256, 0, stream>>>(pos, neg, pn);
  sim_loss_main_kernel<<<dim3(NTILES, BT), 256, 0, stream>>>(nm_h, emb_h, pn,
                                                             numer, denom);
  final_loss_kernel<<<1, 256, 0, stream>>>(numer, denom, (float*)d_out);
}
